// Net_75290776698947
// MI455X (gfx1250) — matrix operand order repaired and checked by reading, not compiled, not run on hardware
//
#include <hip/hip_runtime.h>
#include <hip/hip_bf16.h>

typedef __attribute__((ext_vector_type(16))) _Float16 v16h;
typedef __attribute__((ext_vector_type(8)))  float    v8f;

#define N_NODES 50000
#define N_EDGES 600000
#define IN_F    256
#define EMB     128

// ---------------------------------------------------------------------------
// Degree / normalization
// ---------------------------------------------------------------------------
__global__ __launch_bounds__(256) void deg_init_kernel(float* __restrict__ deg, int n) {
    int i = blockIdx.x * 256 + threadIdx.x;
    if (i < n) deg[i] = 1.0f;   // self loop
}

__global__ __launch_bounds__(256) void deg_accum_kernel(const int* __restrict__ dst,
                                                        float* __restrict__ deg, int nE) {
    int e = blockIdx.x * 256 + threadIdx.x;
    if (e < nE) atomicAdd(&deg[dst[e]], 1.0f);
}

__global__ __launch_bounds__(256) void dinv_kernel(const float* __restrict__ deg,
                                                   float* __restrict__ dinv, int n) {
    int i = blockIdx.x * 256 + threadIdx.x;
    if (i < n) dinv[i] = rsqrtf(deg[i]);
}

// ---------------------------------------------------------------------------
// GEMM: C[M,128] = A[M,K] @ W[K,128]  via v_wmma_f32_16x16x32_f16
// Block = 256 threads = 8 waves. Each block: 16 rows x 128 cols.
// K staged in 128-wide slabs; LDS holds packed f16 pairs so each fragment
// VGPR is one ds_load_b32.
// ---------------------------------------------------------------------------
__global__ __launch_bounds__(256) void gemm_wmma_kernel(const float* __restrict__ A,
                                                        const float* __restrict__ W,
                                                        float* __restrict__ C,
                                                        int M, int K) {
    __shared__ unsigned int As[16 * 64];    //  4 KB: 16 rows  x 64 k-pairs
    __shared__ unsigned int Ws[128 * 64];   // 32 KB: 128 cols x 64 k-pairs

    const int tid  = threadIdx.x;
    const int wave = tid >> 5;          // 0..7 -> output column tile
    const int lane = tid & 31;
    const int half = lane >> 4;         // 0 | 1
    const int ln   = lane & 15;
    const int row0 = blockIdx.x * 16;
    const int n0   = wave * 16;

    v8f c = {};
    const int nStages = K >> 7;         // K/128 (K = 256 or 128)

    for (int st = 0; st < nStages; ++st) {
        const int ks = st << 7;

        // Stage A tile: 16 rows x 64 pairs (float2 -> packed f16 pair)
        for (int idx = tid; idx < 16 * 64; idx += 256) {
            int r = idx >> 6, kp = idx & 63;
            const float* s = A + (size_t)(row0 + r) * K + ks + kp * 2;
            union { _Float16 h[2]; unsigned int u; } p;
            p.h[0] = (_Float16)s[0];
            p.h[1] = (_Float16)s[1];
            As[idx] = p.u;
        }
        // Stage W slab: 128 cols x 64 pairs (W is row-major [K,128])
        for (int idx = tid; idx < 128 * 64; idx += 256) {
            int n = idx >> 6, kp = idx & 63;
            union { _Float16 h[2]; unsigned int u; } p;
            p.h[0] = (_Float16)W[(size_t)(ks + kp * 2)     * 128 + n];
            p.h[1] = (_Float16)W[(size_t)(ks + kp * 2 + 1) * 128 + n];
            Ws[idx] = p.u;
        }
        __syncthreads();

        #pragma unroll
        for (int kc = 0; kc < 4; ++kc) {          // four K=32 chunks per slab
            union { v16h v; unsigned int u[8]; } a, b;
            #pragma unroll
            for (int v = 0; v < 8; ++v) {
                // A 16x32 f16 layout: v<4 -> K=2v+8h ; v>=4 -> K=16+2(v-4)+8h
                int akp = (v < 4) ? (v + 4 * half) : (8 + (v - 4) + 4 * half);
                a.u[v] = As[(ln << 6) + (kc << 4) + akp];
                // B 32x16 f16 layout: same K order as A
                b.u[v] = Ws[((n0 + ln) << 6) + (kc << 4) + akp];
            }
            c = __builtin_amdgcn_wmma_f32_16x16x32_f16(
                    false, a.v, false, b.v, (short)0, c, false, false);
        }
        __syncthreads();
    }

    // C/D 16x16 f32 layout: lane n=L%16, VGPR r -> m = r + 8*(L/16)
    #pragma unroll
    for (int r = 0; r < 8; ++r) {
        int m = row0 + r + 8 * half;
        C[(size_t)m * 128 + n0 + ln] = c[r];
    }
}

// ---------------------------------------------------------------------------
// agg = h * self_norm   (self-loop contribution, also zero-initializes agg)
// ---------------------------------------------------------------------------
__global__ __launch_bounds__(256) void init_agg_kernel(const float* __restrict__ h,
                                                       const float* __restrict__ dinv,
                                                       float* __restrict__ agg, int n) {
    int i = blockIdx.x * 256 + threadIdx.x;
    if (i >= n) return;
    float di = dinv[i >> 7];
    agg[i] = h[i] * di * di;
}

// ---------------------------------------------------------------------------
// Edge scatter: one wave per edge, 32 lanes x float4 = 128 features.
// agg[dst] += h[src] * dinv[src]*dinv[dst]
// ---------------------------------------------------------------------------
__global__ __launch_bounds__(256) void scatter_kernel(const int* __restrict__ src,
                                                      const int* __restrict__ dst,
                                                      const float* __restrict__ dinv,
                                                      const float* __restrict__ h,
                                                      float* __restrict__ agg, int nE) {
    int e = blockIdx.x * 8 + (threadIdx.x >> 5);
    if (e >= nE) return;
    int lane = threadIdx.x & 31;
    int s = src[e], d = dst[e];
    float w = dinv[s] * dinv[d];
    const float4 hv = *(const float4*)(h + (size_t)s * 128 + lane * 4);
    float* out = agg + (size_t)d * 128 + lane * 4;
    atomicAdd(out + 0, hv.x * w);
    atomicAdd(out + 1, hv.y * w);
    atomicAdd(out + 2, hv.z * w);
    atomicAdd(out + 3, hv.w * w);
}

// ---------------------------------------------------------------------------
// o = relu(agg + bias)
// ---------------------------------------------------------------------------
__global__ __launch_bounds__(256) void bias_relu_kernel(const float* __restrict__ agg,
                                                        const float* __restrict__ b,
                                                        float* __restrict__ o, int n) {
    int i = blockIdx.x * 256 + threadIdx.x;
    if (i >= n) return;
    float v = agg[i] + b[i & 127];
    o[i] = v > 0.0f ? v : 0.0f;
}

// ---------------------------------------------------------------------------
// out[i] = dot(o2[i,:], Wl) + bl   — one wave per node, shuffle reduction
// ---------------------------------------------------------------------------
__global__ __launch_bounds__(256) void final_linear_kernel(const float* __restrict__ o,
                                                           const float* __restrict__ Wl,
                                                           const float* __restrict__ bl,
                                                           float* __restrict__ out, int N) {
    int node = blockIdx.x * 8 + (threadIdx.x >> 5);
    if (node >= N) return;
    int lane = threadIdx.x & 31;
    const float4 ov = *(const float4*)(o  + (size_t)node * 128 + lane * 4);
    const float4 wv = *(const float4*)(Wl + lane * 4);
    float s = ov.x * wv.x + ov.y * wv.y + ov.z * wv.z + ov.w * wv.w;
    #pragma unroll
    for (int off = 16; off > 0; off >>= 1) s += __shfl_down(s, off, 32);
    if (lane == 0) out[node] = s + bl[0];
}

// ---------------------------------------------------------------------------
extern "C" void kernel_launch(void* const* d_in, const int* in_sizes, int n_in,
                              void* d_out, int out_size, void* d_ws, size_t ws_size,
                              hipStream_t stream) {
    const float* x   = (const float*)d_in[0];
    const int*   ei  = (const int*)  d_in[1];
    const float* W1  = (const float*)d_in[2];
    const float* b1  = (const float*)d_in[3];
    const float* W2  = (const float*)d_in[4];
    const float* b2  = (const float*)d_in[5];
    const float* Wl  = (const float*)d_in[6];
    const float* bl  = (const float*)d_in[7];
    float* out = (float*)d_out;

    const int E = in_sizes[1] / 2;      // edge_index is [2, E]
    const int*   src = ei;
    const int*   dst = ei + E;

    // Workspace layout (floats): deg | dinv | bufA | bufB
    float* ws   = (float*)d_ws;
    float* deg  = ws;
    float* dinv = ws + N_NODES;
    float* bufA = ws + 2 * N_NODES;                       // 50000*128 floats
    float* bufB = bufA + (size_t)N_NODES * EMB;

    const int NF = N_NODES * EMB;                         // 6.4M elements
    const int nTiles = N_NODES / 16;                      // 3125 (exact)

    // 1) degree + normalization
    deg_init_kernel <<<(N_NODES + 255) / 256, 256, 0, stream>>>(deg, N_NODES);
    deg_accum_kernel<<<(E + 255) / 256,       256, 0, stream>>>(dst, deg, E);
    dinv_kernel     <<<(N_NODES + 255) / 256, 256, 0, stream>>>(deg, dinv, N_NODES);

    // 2) layer 1: h1 = x @ W1  (bufA)
    gemm_wmma_kernel<<<nTiles, 256, 0, stream>>>(x, W1, bufA, N_NODES, IN_F);
    // agg1 = h1*self_norm (bufB); += edge messages; o1 = relu(agg1+b1) -> bufA
    init_agg_kernel <<<(NF + 255) / 256, 256, 0, stream>>>(bufA, dinv, bufB, NF);
    scatter_kernel  <<<(E + 7) / 8,      256, 0, stream>>>(src, dst, dinv, bufA, bufB, E);
    bias_relu_kernel<<<(NF + 255) / 256, 256, 0, stream>>>(bufB, b1, bufA, NF);

    // 3) layer 2: h2 = o1 @ W2 (bufB)
    gemm_wmma_kernel<<<nTiles, 256, 0, stream>>>(bufA, W2, bufB, N_NODES, EMB);
    init_agg_kernel <<<(NF + 255) / 256, 256, 0, stream>>>(bufB, dinv, bufA, NF);
    scatter_kernel  <<<(E + 7) / 8,      256, 0, stream>>>(src, dst, dinv, bufB, bufA, E);
    bias_relu_kernel<<<(NF + 255) / 256, 256, 0, stream>>>(bufA, b2, bufB, NF);

    // 4) out = o2 @ Wl + bl
    final_linear_kernel<<<(N_NODES + 7) / 8, 256, 0, stream>>>(bufB, Wl, bl, out, N_NODES);

    (void)n_in; (void)out_size; (void)ws_size;
}
